// LISTA_3221225472658
// MI455X (gfx1250) — compile-verified, matrix operand order checked by reading, not run on hardware
//
#include <hip/hip_runtime.h>

// ---------------------------------------------------------------------------
// Fused LISTA on MI455X (gfx1250): x_{i+1} = shrink(x + g*(Ay - x @ B^T), eta)
// All 16 iterations fused; x lives in LDS, accumulators in VGPRs,
// matmuls via v_wmma_f32_16x16x32_bf16 (wave32).
// ---------------------------------------------------------------------------

typedef __attribute__((ext_vector_type(16))) __bf16 v16bf;
typedef __attribute__((ext_vector_type(8)))  float  v8f;

#define STRD    520   // LDS row stride in bf16 elements (512 + 8 pad, 16B-aligned rows, bank-skewed)
#define BM      64    // batch rows per workgroup
#define LAMBD_C 1.0f

union FragU { uint4 q[2]; v16bf v; };

__device__ __forceinline__ unsigned short f2bf(float f) {
  union { float f; unsigned u; } c; c.f = f;
  unsigned u = c.u + 0x7FFFu + ((c.u >> 16) & 1u);   // round-to-nearest-even
  return (unsigned short)(u >> 16);
}
__device__ __forceinline__ float bf2f(unsigned short h) {
  union { unsigned u; float f; } c; c.u = ((unsigned)h) << 16;
  return c.f;
}
__device__ __forceinline__ float shrinkf(float v, float eta) {
  float t = fabsf(v) - eta * LAMBD_C;
  return t > 0.0f ? copysignf(t, v) : 0.0f;
}

// ---------------------------------------------------------------------------
// Weight pre-pack: f32 row-major [N,K] -> bf16 WMMA B-operand fragments.
// Fragment layout (assumed mirror of documented 16-bit A layout, ISA 7.12.2):
//   lane[0..15] = column n, lane>>4 selects K half-group; 16 bf16 per lane,
//   element e: vgpr g=e/2, pair p=e%2,
//   k_local = (g<4 ? 2g+p : 16+2(g-4)+p) + (lane>=16 ? 8 : 0)
// Stored lane-major: frag base + lane*16 elements -> hot loop does 2x b128.
// ---------------------------------------------------------------------------
__device__ __forceinline__ int klocal_of(int e, int hi) {
  int g = e >> 1, p = e & 1;
  return (g < 4 ? 2 * g + p : 16 + 2 * (g - 4) + p) + hi * 8;
}

__global__ void pack_a_kernel(const float* __restrict__ Aw, unsigned short* __restrict__ out) {
  int f    = blockIdx.x * 256 + threadIdx.x;   // 512*256 elements
  int frag = f >> 9;
  int e16  = f & 511;
  int lane = e16 >> 4;
  int e    = e16 & 15;
  int nt   = frag >> 3;          // N tile over 512 (Ay columns)
  int kt   = frag & 7;           // K tile over 256
  int n    = nt * 16 + (lane & 15);
  int k    = kt * 32 + klocal_of(e, lane >> 4);
  out[f] = f2bf(Aw[n * 256 + k]);
}

__global__ void pack_b_kernel(const float* __restrict__ Bw, unsigned short* __restrict__ out) {
  int f    = blockIdx.x * 256 + threadIdx.x;   // 512*512 elements
  int frag = f >> 9;
  int e16  = f & 511;
  int lane = e16 >> 4;
  int e    = e16 & 15;
  int nt   = frag >> 4;          // N tile over 512
  int kt   = frag & 15;          // K tile over 512
  int n    = nt * 16 + (lane & 15);
  int k    = kt * 32 + klocal_of(e, lane >> 4);
  out[f] = f2bf(Bw[n * 512 + k]);
}

// ---------------------------------------------------------------------------
// Main fused kernel. 256 threads = 8 waves. Wave w owns all 64 rows x the
// 64-column slab [w*64, w*64+64) => 4x4 tiles of 16x16 f32 accumulators.
// ---------------------------------------------------------------------------
__global__ __launch_bounds__(256)
void lista_fused_kernel(const float* __restrict__ y,
                        const float* __restrict__ etas,
                        const float* __restrict__ gammas,
                        const unsigned short* __restrict__ Ap,
                        const unsigned short* __restrict__ Bp,
                        float* __restrict__ out, int T) {
  __shared__ __align__(16) unsigned short lx [BM * STRD];  // x   (bf16), also y staging
  __shared__ __align__(16) unsigned short lay[BM * STRD];  // Ay  (bf16)

  const int tid  = threadIdx.x;
  const int w    = tid >> 5;
  const int lane = tid & 31;
  const int lo   = lane & 15;
  const int hi   = lane >> 4;
  const long rowbase = (long)blockIdx.x * BM;

  // ---- Phase A: stage y tile (64 x 256) into LDS as bf16 (coalesced) ----
  for (int i = tid; i < BM * 256; i += 256) {
    int r = i >> 8, c = i & 255;
    lx[r * STRD + c] = f2bf(y[(rowbase + r) * 256 + c]);
  }
  __syncthreads();

  v8f acc[4][4];
  const v8f vzero = {0.f, 0.f, 0.f, 0.f, 0.f, 0.f, 0.f, 0.f};

  // ---- Phase B: Ay = y @ A_w^T (K = 256, 8 k-steps) ----
#pragma unroll
  for (int im = 0; im < 4; ++im)
#pragma unroll
    for (int jn = 0; jn < 4; ++jn) acc[im][jn] = vzero;

  for (int kt = 0; kt < 8; ++kt) {
    FragU A[4];
#pragma unroll
    for (int im = 0; im < 4; ++im) {
      const unsigned short* pa = &lx[(im * 16 + lo) * STRD + kt * 32 + hi * 8];
      A[im].q[0] = *(const uint4*)(pa);
      A[im].q[1] = *(const uint4*)(pa + 16);   // k_local + 16
    }
#pragma unroll
    for (int jn = 0; jn < 4; ++jn) {
      const int nt = w * 4 + jn;
      const unsigned short* pb = Ap + (((nt << 3) + kt) << 9) + (lane << 4);
      FragU B;
      B.q[0] = *(const uint4*)(pb);
      B.q[1] = *(const uint4*)(pb + 8);
#pragma unroll
      for (int im = 0; im < 4; ++im)
        acc[im][jn] = __builtin_amdgcn_wmma_f32_16x16x32_bf16(
            false, A[im].v, false, B.v, (short)0, acc[im][jn], false, false);
    }
  }
  __syncthreads();   // all waves done reading staged y from lx

  // ---- x0 = shrink(g0 * Ay, eta0); stash Ay (bf16) ----
  {
    const float g = gammas[0], eta = etas[0];
#pragma unroll
    for (int im = 0; im < 4; ++im)
#pragma unroll
      for (int jn = 0; jn < 4; ++jn) {
        const int c = w * 64 + jn * 16 + lo;
#pragma unroll
        for (int r = 0; r < 8; ++r) {
          const int row = im * 16 + hi * 8 + r;      // C-matrix VGPR layout
          const float ay = acc[im][jn][r];
          lay[row * STRD + c] = f2bf(ay);
          lx [row * STRD + c] = f2bf(shrinkf(g * ay, eta));
        }
      }
  }
  __syncthreads();

  // ---- Phase C: T fused iterations, K = 512 (16 k-steps) ----
  for (int it = 1; it <= T; ++it) {
    const float g = gammas[it], eta = etas[it];
#pragma unroll
    for (int im = 0; im < 4; ++im)
#pragma unroll
      for (int jn = 0; jn < 4; ++jn) acc[im][jn] = vzero;

    for (int kt = 0; kt < 16; ++kt) {
      FragU A[4];
#pragma unroll
      for (int im = 0; im < 4; ++im) {
        const unsigned short* pa = &lx[(im * 16 + lo) * STRD + kt * 32 + hi * 8];
        A[im].q[0] = *(const uint4*)(pa);
        A[im].q[1] = *(const uint4*)(pa + 16);
      }
#pragma unroll
      for (int jn = 0; jn < 4; ++jn) {
        const int nt = w * 4 + jn;
        const unsigned short* pb = Bp + (((nt << 4) + kt) << 9) + (lane << 4);
        FragU B;
        B.q[0] = *(const uint4*)(pb);
        B.q[1] = *(const uint4*)(pb + 8);
#pragma unroll
        for (int im = 0; im < 4; ++im)
          acc[im][jn] = __builtin_amdgcn_wmma_f32_16x16x32_bf16(
              false, A[im].v, false, B.v, (short)0, acc[im][jn], false, false);
      }
    }
    __syncthreads();   // all GEMM reads of old x complete before rewrite

    if (it < T) {
#pragma unroll
      for (int im = 0; im < 4; ++im)
#pragma unroll
        for (int jn = 0; jn < 4; ++jn) {
          const int c = w * 64 + jn * 16 + lo;
#pragma unroll
          for (int r = 0; r < 8; ++r) {
            const int row = im * 16 + hi * 8 + r;
            const float xold = bf2f(lx [row * STRD + c]);
            const float ayv  = bf2f(lay[row * STRD + c]);
            const float v = xold + g * (ayv - acc[im][jn][r]);
            lx[row * STRD + c] = f2bf(shrinkf(v, eta));
          }
        }
      __syncthreads();
    } else {
      // final iteration: store f32 result straight to HBM
#pragma unroll
      for (int im = 0; im < 4; ++im)
#pragma unroll
        for (int jn = 0; jn < 4; ++jn) {
          const int c = w * 64 + jn * 16 + lo;
#pragma unroll
          for (int r = 0; r < 8; ++r) {
            const int row = im * 16 + hi * 8 + r;
            const float xold = bf2f(lx [row * STRD + c]);
            const float ayv  = bf2f(lay[row * STRD + c]);
            const float v = xold + g * (ayv - acc[im][jn][r]);
            out[(rowbase + row) * 512 + c] = shrinkf(v, eta);
          }
        }
    }
  }
}

// ---------------------------------------------------------------------------
extern "C" void kernel_launch(void* const* d_in, const int* in_sizes, int n_in,
                              void* d_out, int out_size, void* d_ws, size_t ws_size,
                              hipStream_t stream) {
  const float* y     = (const float*)d_in[0];   // [65536, 256]
  const float* Aw    = (const float*)d_in[1];   // [512, 256]
  const float* Bw    = (const float*)d_in[2];   // [512, 512]
  const float* etas  = (const float*)d_in[3];   // [T+1]
  const float* gam   = (const float*)d_in[4];   // [T+1]
  float*       outp  = (float*)d_out;           // [65536, 512]

  unsigned short* Ap = (unsigned short*)d_ws;         // 512*256 bf16 = 256 KB
  unsigned short* Bpk = Ap + 512 * 256;               // 512*512 bf16 = 512 KB

  const int T     = in_sizes[3] - 1;                  // 16
  const int batch = in_sizes[0] / 256;                // 65536
  const int nwg   = batch / BM;                       // 1024

  hipLaunchKernelGGL(pack_a_kernel, dim3((512 * 256) / 256), dim3(256), 0, stream, Aw, Ap);
  hipLaunchKernelGGL(pack_b_kernel, dim3((512 * 512) / 256), dim3(256), 0, stream, Bw, Bpk);
  hipLaunchKernelGGL(lista_fused_kernel, dim3(nwg), dim3(256), 0, stream,
                     y, etas, gam, Ap, Bpk, outp, T);
}
